// Model_44049184588238
// MI455X (gfx1250) — matrix operand order repaired and checked by reading, not compiled, not run on hardware
//
#include <hip/hip_runtime.h>
#include <cstdint>
#include <cstddef>

// ---------------- types ----------------
typedef __attribute__((ext_vector_type(16))) __bf16 v16bf;
typedef __attribute__((ext_vector_type(8)))  __bf16 v8bf;
typedef __attribute__((ext_vector_type(8)))  float  v8f;
typedef __attribute__((ext_vector_type(4)))  unsigned int v4u;
typedef __attribute__((ext_vector_type(8)))  int v8i;
typedef __attribute__((ext_vector_type(4)))  int v4i;

#if __has_include(<hip/amd_detail/amd_gfx1250_TDM.h>)
#define TDM_SIX_ARGS 1
#endif

// ---------------- problem constants ----------------
constexpr int NATOM  = 10240;     // B*M = 256*40
constexpr int NEDGE  = 81920;     // E
constexpr int NPAIR  = 199680;    // 256 * 780
constexpr int DIM    = 256;       // D
constexpr int HID    = 512;       // H
constexpr int VCH    = 16;        // V
constexpr int KCODE  = 4096;      // K
constexpr int NTYPE  = 10;
constexpr int NCH    = 11;
constexpr int NBOND  = 5;
constexpr int RBFD   = 32;
constexpr int LAYERS = 4;
constexpr int MIND   = DIM + RBFD + 1 + VCH;   // 305
constexpr int BONDK  = DIM + 1;                // 257
constexpr int PCHUNK = 49920;                  // NPAIR/4
constexpr int NCHUNK = 4;
constexpr float RBF_DMAX = 10.0f;

// ---------------- GEMM tile config ----------------
constexpr int BM = 128, BN = 64, BK = 32;
constexpr int LDA_S = 40;   // halves per LDS A row: 32 + 8 pad (80B, 16B-aligned, skewed)
constexpr int LDB_S = 72;   // halves per LDS B row: 64 + 8 pad  ([k][n] layout)

__device__ __forceinline__ v8f zero8() {
  v8f z;
#pragma unroll
  for (int i = 0; i < 8; ++i) z[i] = 0.0f;
  return z;
}

union FragBF { v16bf v; v8bf h[2]; };

// generic pointer (to __shared__) -> 32-bit LDS byte offset for the TDM descriptor
__device__ __forceinline__ unsigned lds_off_u32(const void* p) {
  return (unsigned)(unsigned long long)(const __attribute__((address_space(3))) void*)p;
}

// ---------------- Tensor Data Mover: async 2D bf16 tile -> LDS ----------------
// Loads a tile_d1 x tile_d0 (rows x contiguous-elements) bf16 tile starting at
// gptr (row stride = stride_e elements) into LDS at lds_off, with LDS padding
// pad_amount_code+1 DWORDs every 2^(pad_interval_code+1) DWORDs.
// tensor_d0/d1 = remaining in-bounds extent -> hardware zero-fills the OOB tail.
__device__ __forceinline__ void tdm_load_2d_bf16(unsigned lds_off, const void* gptr,
                                                 int tile_d0, int tile_d1,
                                                 long long tensor_d0, long long tensor_d1,
                                                 long long stride_e,
                                                 int pad_interval_code, int pad_amount_code) {
  unsigned long long ga = (unsigned long long)gptr;
  v4u g0;
  g0[0] = 1u;                                        // count=1 (valid), user mode
  g0[1] = lds_off;                                   // lds_addr
  g0[2] = (unsigned)(ga & 0xFFFFFFFFu);              // global_addr[31:0]
  g0[3] = (unsigned)((ga >> 32) & 0x01FFFFFFu) | (2u << 30);  // addr[56:32] | type=2
  unsigned td0 = (unsigned)tensor_d0, td1 = (unsigned)tensor_d1;
  unsigned long long st = (unsigned long long)stride_e;
  unsigned d0 = (1u << 16)                           // data_size = 2 bytes
              | (1u << 20)                           // pad_enable
              | ((unsigned)pad_interval_code << 22)
              | ((unsigned)pad_amount_code << 25);
  v8i g1;
  g1[0] = (int)d0;
  g1[1] = (int)((td0 & 0xFFFFu) << 16);                          // tensor_dim0[15:0]
  g1[2] = (int)(((td0 >> 16) & 0xFFFFu) | ((td1 & 0xFFFFu) << 16));
  g1[3] = (int)(((td1 >> 16) & 0xFFFFu) | ((unsigned)tile_d0 << 16));  // tile_dim0
  g1[4] = (int)((unsigned)tile_d1 & 0xFFFFu);                    // tile_dim1 (dim2=0)
  g1[5] = (int)(unsigned)(st & 0xFFFFFFFFu);                     // dim0_stride[31:0]
  g1[6] = (int)(unsigned)((st >> 32) & 0xFFFFu);                 // dim0_stride[47:32]
  g1[7] = 0;
  v4i z4 = {0, 0, 0, 0};
#ifdef TDM_SIX_ARGS
  v8i z8 = {0, 0, 0, 0, 0, 0, 0, 0};
  __builtin_amdgcn_tensor_load_to_lds(g0, g1, z4, z4, z8, 0);
#else
  __builtin_amdgcn_tensor_load_to_lds(g0, g1, z4, z4, 0);
#endif
}

// ---------------- generic bf16 WMMA GEMM  C = epi(A*B + bias) ----------------
// A: (M,K) row-major bf16.  B: (K,N) row-major bf16.  Cf fp32 / Cbf bf16 optional.
// TDM double-buffered LDS staging; EPI: 0=none, 1=relu, 2=sigmoid
template <int EPI>
__global__ __launch_bounds__(256)
void gemm_bf16_wmma(const __bf16* __restrict__ A, const __bf16* __restrict__ B,
                    const float* __restrict__ bias,
                    float* __restrict__ Cf, __bf16* __restrict__ Cbf,
                    int M, int N, int K, int ldc) {
  __shared__ alignas(16) __bf16 sA[2][BM * LDA_S];   // [row][k] skewed
  __shared__ alignas(16) __bf16 sB[2][BK * LDB_S];   // [k][n]  skewed

  const int tid  = threadIdx.x;
  const int wave = tid >> 5, lane = tid & 31;
  const int wm = wave >> 1, wn = wave & 1;            // 4x2 wave grid
  const int bm0 = blockIdx.x * BM, bn0 = blockIdx.y * BN;
  const int khalf = lane >> 4, lr = lane & 15;
  const int nk = (K + BK - 1) / BK;

  v8f acc[2][2];
  acc[0][0] = zero8(); acc[0][1] = zero8(); acc[1][0] = zero8(); acc[1][1] = zero8();

  // wave 0 drives the Tensor Data Mover for both tiles of a k-step
  auto issue = [&](int kk, int buf) {
    const int k0 = kk * BK;
    // A tile: BM rows x BK elems; LDS row stride 40 halves (pad 4 DW every 16 DW)
    tdm_load_2d_bf16(lds_off_u32(&sA[buf][0]), A + (size_t)bm0 * K + k0,
                     BK, BM, (long long)(K - k0), (long long)(M - bm0),
                     (long long)K, /*interval 2^(3+1)=16DW*/ 3, /*4 DW*/ 3);
    // B tile: BK rows x BN elems; LDS row stride 72 halves (pad 4 DW every 32 DW)
    tdm_load_2d_bf16(lds_off_u32(&sB[buf][0]), B + (size_t)k0 * N + bn0,
                     BN, BK, (long long)(N - bn0), (long long)(K - k0),
                     (long long)N, /*interval 2^(4+1)=32DW*/ 4, /*4 DW*/ 3);
  };

  if (wave == 0) issue(0, 0);

  for (int kk = 0; kk < nk; ++kk) {
    const int buf = kk & 1;
    if (wave == 0) __builtin_amdgcn_s_wait_tensorcnt(0);
    __syncthreads();                         // current buffer's DMA complete
    if (wave == 0 && kk + 1 < nk) issue(kk + 1, buf ^ 1);  // prefetch next tile

    // ---- fragments (wave32 layouts) ----
    FragBF fa[2], fb[2];
#pragma unroll
    for (int mi = 0; mi < 2; ++mi) {
      const __bf16* p = &sA[buf][(wm * 32 + mi * 16 + lr) * LDA_S];
      fa[mi].h[0] = *(const v8bf*)(p + 16 * khalf);       // K = 16*khalf + 0..7
      fa[mi].h[1] = *(const v8bf*)(p + 16 * khalf + 8);   // K = 16*khalf + 8..15
    }
#pragma unroll
    for (int ni = 0; ni < 2; ++ni) {
      const int n = wn * 32 + ni * 16 + lr;
      const __bf16* pb = &sB[buf][(16 * khalf) * LDB_S + n];
#pragma unroll
      for (int e = 0; e < 16; ++e)                        // element e: K = 16*khalf + e
        fb[ni].v[e] = pb[e * LDB_S];
    }
#pragma unroll
    for (int mi = 0; mi < 2; ++mi)
#pragma unroll
      for (int ni = 0; ni < 2; ++ni)
        acc[mi][ni] = __builtin_amdgcn_wmma_f32_16x16x32_bf16(
            false, fa[mi].v, false, fb[ni].v, (short)0, acc[mi][ni], false, false);
    __syncthreads();
  }

  // ---- epilogue: C lane = col, 8 rows per lane (rows 0-7 / 8-15 by lane half)
#pragma unroll
  for (int mi = 0; mi < 2; ++mi)
#pragma unroll
    for (int ni = 0; ni < 2; ++ni) {
      const int gn = bn0 + wn * 32 + ni * 16 + lr;
      if (gn >= N) continue;
      const float bv = bias ? bias[gn] : 0.0f;
#pragma unroll
      for (int j = 0; j < 8; ++j) {
        const int gm = bm0 + wm * 32 + mi * 16 + khalf * 8 + j;
        if (gm >= M) continue;
        float x = acc[mi][ni][j] + bv;
        if (EPI == 1) x = fmaxf(x, 0.0f);
        if (EPI == 2) x = 1.0f / (1.0f + __expf(-x));
        if (Cf)  Cf[(size_t)gm * ldc + gn] = x;
        if (Cbf) Cbf[(size_t)gm * ldc + gn] = (__bf16)x;
      }
    }
}

// ---------------- fused VQ nearest-code search (WMMA, direct-global frags) ----
// per block: 16 atoms; scan all codes; argmin of (||c||^2 - 2 z.c) per atom.
__global__ __launch_bounds__(256)
void vq_argmin_kernel(const __bf16* __restrict__ zbf, const __bf16* __restrict__ cbf,
                      const float* __restrict__ cnorm, int* __restrict__ outIdx) {
  __shared__ unsigned long long best[16];
  const int tid = threadIdx.x, wave = tid >> 5, lane = tid & 31;
  const int khalf = lane >> 4, lr = lane & 15;
  const int atom0 = blockIdx.x * 16;
  if (tid < 16) best[tid] = 0xFFFFFFFFFFFFFFFFull;
  __syncthreads();

  for (int c0 = wave * 16; c0 < KCODE; c0 += 128) {
    v8f acc = zero8();
    FragBF fa, fb;
    for (int d0 = 0; d0 < DIM; d0 += 32) {
      const __bf16* pa = zbf + (size_t)(atom0 + lr) * DIM + d0;
      fa.h[0] = *(const v8bf*)(pa + 16 * khalf);
      fa.h[1] = *(const v8bf*)(pa + 16 * khalf + 8);
      const __bf16* pb = cbf + (size_t)(c0 + lr) * DIM + d0 + 16 * khalf;
      fb.h[0] = *(const v8bf*)(pb);
      fb.h[1] = *(const v8bf*)(pb + 8);
      acc = __builtin_amdgcn_wmma_f32_16x16x32_bf16(
          false, fa.v, false, fb.v, (short)0, acc, false, false);
    }
    const int code = c0 + lr;
    const float cn = cnorm[code];
#pragma unroll
    for (int j = 0; j < 8; ++j) {
      const int m = khalf * 8 + j;
      float val = cn - 2.0f * acc[j];
      unsigned u = __float_as_uint(val);
      u = (u & 0x80000000u) ? ~u : (u | 0x80000000u);   // orderable
      unsigned long long key = ((unsigned long long)u << 32) | (unsigned)code;
      atomicMin(&best[m], key);
    }
  }
  __syncthreads();
  if (tid < 16) outIdx[atom0 + tid] = (int)(best[tid] & 0xFFFFFFFFu);
}

// ---------------- element-wise / scatter / loss kernels ----------------
__global__ void k_zero_f32(float* p, long long n) {
  for (long long g = (long long)blockIdx.x * blockDim.x + threadIdx.x; g < n;
       g += (long long)gridDim.x * blockDim.x) p[g] = 0.0f;
}
__global__ void k_zero_i32(int* p, long long n) {
  for (long long g = (long long)blockIdx.x * blockDim.x + threadIdx.x; g < n;
       g += (long long)gridDim.x * blockDim.x) p[g] = 0;
}
__global__ void k_f32_bf16(const float* __restrict__ s, __bf16* __restrict__ d, long long n) {
  for (long long g = (long long)blockIdx.x * blockDim.x + threadIdx.x; g < n;
       g += (long long)gridDim.x * blockDim.x) d[g] = (__bf16)s[g];
}

__global__ void k_embed(const int* __restrict__ types, const int* __restrict__ charges,
                        const float* __restrict__ w0, const float* __restrict__ b0,
                        __bf16* __restrict__ Hemb) {
  long long n = (long long)NATOM * HID;
  for (long long g = (long long)blockIdx.x * blockDim.x + threadIdx.x; g < n;
       g += (long long)gridDim.x * blockDim.x) {
    int a = (int)(g / HID), h = (int)(g % HID);
    float x = w0[types[a] * HID + h] + w0[(NTYPE + charges[a] + 5) * HID + h] + b0[h];
    Hemb[g] = (__bf16)fmaxf(x, 0.0f);
  }
}

__global__ void k_geom(const float* __restrict__ coords, const int* __restrict__ src,
                       const int* __restrict__ dst, float* __restrict__ xdiff,
                       float* __restrict__ dlen) {
  int e = blockIdx.x * blockDim.x + threadIdx.x;
  if (e >= NEDGE) return;
  int s = src[e], d = dst[e];
  float dx = coords[s * 3 + 0] - coords[d * 3 + 0];
  float dy = coords[s * 3 + 1] - coords[d * 3 + 1];
  float dz = coords[s * 3 + 2] - coords[d * 3 + 2];
  float nn = sqrtf(fmaxf(dx * dx + dy * dy + dz * dz, 1e-8f));
  dlen[e] = nn;
  xdiff[e * 3 + 0] = dx / nn; xdiff[e * 3 + 1] = dy / nn; xdiff[e * 3 + 2] = dz / nn;
}

__global__ void k_deg(const int* __restrict__ dst, float* __restrict__ deg) {
  int e = blockIdx.x * blockDim.x + threadIdx.x;
  if (e < NEDGE) atomicAdd(&deg[dst[e]], 1.0f);
}

// fills m_in cols [0..288]: s[src] | rbf(d) | bond ;  cols 289..304 filled by k_vh
__global__ void k_build_min(const float* __restrict__ s, const float* __restrict__ dlen,
                            const int* __restrict__ src, const int* __restrict__ ebond,
                            __bf16* __restrict__ m_in) {
  long long n = (long long)NEDGE * MIND;
  const float sigma = RBF_DMAX / RBFD;                 // 0.3125
  const float mustep = RBF_DMAX / (RBFD - 1);
  for (long long g = (long long)blockIdx.x * blockDim.x + threadIdx.x; g < n;
       g += (long long)gridDim.x * blockDim.x) {
    int e = (int)(g / MIND), c = (int)(g % MIND);
    if (c < DIM) {
      m_in[g] = (__bf16)s[(size_t)src[e] * DIM + c];
    } else if (c < DIM + RBFD) {
      int j = c - DIM;
      float t = (dlen[e] - j * mustep) / sigma;
      m_in[g] = (__bf16)__expf(-t * t);
    } else if (c == DIM + RBFD) {
      m_in[g] = (__bf16)(float)ebond[e];
    }
    // c >= 289: vn, written by k_vh
  }
}

// vh[e][u][:] = sum_v v_in[e][v][:] * Wh[v][u] ; vn -> m_in col 289+u
__global__ void k_vh(const float* __restrict__ v, const float* __restrict__ xdiff,
                     const int* __restrict__ src, const float* __restrict__ Wh_i,
                     float* __restrict__ vh, __bf16* __restrict__ m_in) {
  long long n = (long long)NEDGE * VCH;
  for (long long g = (long long)blockIdx.x * blockDim.x + threadIdx.x; g < n;
       g += (long long)gridDim.x * blockDim.x) {
    int e = (int)(g / VCH), u = (int)(g % VCH);
    const float* vs = v + (size_t)src[e] * (VCH * 3);
    float a0 = 0.f, a1 = 0.f, a2 = 0.f;
#pragma unroll
    for (int vv = 0; vv < VCH; ++vv) {
      float w = Wh_i[vv * VCH + u];
      a0 += vs[vv * 3 + 0] * w; a1 += vs[vv * 3 + 1] * w; a2 += vs[vv * 3 + 2] * w;
    }
    {
      float w = Wh_i[VCH * VCH + u];  // row 16 = x_diff
      a0 += xdiff[e * 3 + 0] * w; a1 += xdiff[e * 3 + 1] * w; a2 += xdiff[e * 3 + 2] * w;
    }
    vh[(size_t)e * (VCH * 3) + u * 3 + 0] = a0;
    vh[(size_t)e * (VCH * 3) + u * 3 + 1] = a1;
    vh[(size_t)e * (VCH * 3) + u * 3 + 2] = a2;
    float vn = sqrtf(fmaxf(a0 * a0 + a1 * a1 + a2 * a2, 1e-8f));
    m_in[(size_t)e * MIND + (DIM + RBFD + 1) + u] = (__bf16)vn;
  }
}

__global__ void k_sscatter(const float* __restrict__ s_m, const int* __restrict__ dst,
                           const float* __restrict__ deg, float* __restrict__ s) {
  long long n = (long long)NEDGE * DIM;
  for (long long g = (long long)blockIdx.x * blockDim.x + threadIdx.x; g < n;
       g += (long long)gridDim.x * blockDim.x) {
    int e = (int)(g / DIM), c = (int)(g % DIM);
    int d = dst[e];
    float inv = 1.0f / fmaxf(deg[d], 1.0f);
    atomicAdd(&s[(size_t)d * DIM + c], s_m[g] * inv);
  }
}

__global__ void k_vscatter(const float* __restrict__ vh, const float* __restrict__ gate,
                           const float* __restrict__ Wv_i, const int* __restrict__ dst,
                           const float* __restrict__ deg, float* __restrict__ v) {
  long long n = (long long)NEDGE * VCH;
  for (long long g = (long long)blockIdx.x * blockDim.x + threadIdx.x; g < n;
       g += (long long)gridDim.x * blockDim.x) {
    int e = (int)(g / VCH), u = (int)(g % VCH);
    const float* ve = vh + (size_t)e * (VCH * 3);
    float a0 = 0.f, a1 = 0.f, a2 = 0.f;
#pragma unroll
    for (int vv = 0; vv < VCH; ++vv) {
      float w = Wv_i[vv * VCH + u];
      a0 += ve[vv * 3 + 0] * w; a1 += ve[vv * 3 + 1] * w; a2 += ve[vv * 3 + 2] * w;
    }
    float gt = gate[(size_t)e * VCH + u];
    int d = dst[e];
    float inv = gt / fmaxf(deg[d], 1.0f);
    atomicAdd(&v[(size_t)d * (VCH * 3) + u * 3 + 0], a0 * inv);
    atomicAdd(&v[(size_t)d * (VCH * 3) + u * 3 + 1], a1 * inv);
    atomicAdd(&v[(size_t)d * (VCH * 3) + u * 3 + 2], a2 * inv);
  }
}

__global__ void k_cnorm(const float* __restrict__ cb, float* __restrict__ cnorm) {
  int k = blockIdx.x * blockDim.x + threadIdx.x;
  if (k >= KCODE) return;
  float s = 0.f;
  for (int d = 0; d < DIM; ++d) { float x = cb[(size_t)k * DIM + d]; s += x * x; }
  cnorm[k] = s;
}

// q = codebook[idx]; qbf; accumulate sum((q - z)^2) into acc[0]
__global__ void k_gather_q(const float* __restrict__ cb, const int* __restrict__ idx,
                           const float* __restrict__ s, float* __restrict__ q,
                           __bf16* __restrict__ qbf, float* __restrict__ acc) {
  long long n = (long long)NATOM * DIM;
  float local = 0.f;
  for (long long g = (long long)blockIdx.x * blockDim.x + threadIdx.x; g < n;
       g += (long long)gridDim.x * blockDim.x) {
    int a = (int)(g / DIM), c = (int)(g % DIM);
    float qv = cb[(size_t)idx[a] * DIM + c];
    q[g] = qv;
    qbf[g] = (__bf16)qv;
    float d = qv - s[g];
    local += d * d;
  }
  atomicAdd(&acc[0], local);
}

__global__ void k_hist(const int* __restrict__ idx, int* __restrict__ cnt) {
  int a = blockIdx.x * blockDim.x + threadIdx.x;
  if (a < NATOM) atomicAdd(&cnt[idx[a]], 1);
}

__global__ void k_split(const float* __restrict__ logits, float* __restrict__ at_out,
                        float* __restrict__ ch_out) {
  long long n = (long long)NATOM * (NTYPE + NCH);
  for (long long g = (long long)blockIdx.x * blockDim.x + threadIdx.x; g < n;
       g += (long long)gridDim.x * blockDim.x) {
    int r = (int)(g / (NTYPE + NCH)), c = (int)(g % (NTYPE + NCH));
    float x = logits[g];
    if (c < NTYPE) at_out[(size_t)r * NTYPE + c] = x;
    else           ch_out[(size_t)r * NCH + (c - NTYPE)] = x;
  }
}

__global__ void k_bondX(const float* __restrict__ q, const float* __restrict__ coords,
                        const int* __restrict__ pairs, __bf16* __restrict__ X,
                        int p0, int pc) {
  long long n = (long long)pc * BONDK;
  for (long long g = (long long)blockIdx.x * blockDim.x + threadIdx.x; g < n;
       g += (long long)gridDim.x * blockDim.x) {
    int pr = (int)(g / BONDK), c = (int)(g % BONDK);
    int p = p0 + pr;
    int i0 = pairs[p * 2 + 0], i1 = pairs[p * 2 + 1];
    if (c < DIM) {
      X[g] = (__bf16)(q[(size_t)i0 * DIM + c] + q[(size_t)i1 * DIM + c]);
    } else {
      float dx = coords[i0 * 3 + 0] - coords[i1 * 3 + 0];
      float dy = coords[i0 * 3 + 1] - coords[i1 * 3 + 1];
      float dz = coords[i0 * 3 + 2] - coords[i1 * 3 + 2];
      X[g] = (__bf16)sqrtf(fmaxf(dx * dx + dy * dy + dz * dz, 1e-8f));
    }
  }
}

__global__ void k_ce(const float* __restrict__ logits, int rows, int ncls,
                     const int* __restrict__ labels, int ladd, float* __restrict__ accslot) {
  float local = 0.f;
  for (int r = blockIdx.x * blockDim.x + threadIdx.x; r < rows;
       r += gridDim.x * blockDim.x) {
    const float* row = logits + (size_t)r * ncls;
    float mx = row[0];
    for (int c = 1; c < ncls; ++c) mx = fmaxf(mx, row[c]);
    float se = 0.f;
    for (int c = 0; c < ncls; ++c) se += __expf(row[c] - mx);
    float lse = mx + logf(se);
    local += lse - row[labels[r] + ladd];
  }
  atomicAdd(accslot, local);
}

__global__ void k_final(const float* __restrict__ acc, const int* __restrict__ cnt,
                        float* __restrict__ out, int out_size) {
  if (blockIdx.x != 0 || threadIdx.x != 0) return;
  float msq = acc[0] / (float)((long long)NATOM * DIM);
  float loss = 1.25f * msq + acc[1] / (float)NATOM + acc[2] / (float)NATOM +
               acc[3] / (float)NPAIR;
  float H = 0.f;
  for (int k = 0; k < KCODE; ++k) {
    float a = (float)cnt[k] / (float)NATOM;
    H += a * logf(a + 1e-10f);
  }
  out[0] = loss;
  out[out_size - 1] = __expf(-H);
}

// ---------------- host-side helpers ----------------
static void launch_gemm(int epi, const __bf16* A, const __bf16* B, const float* bias,
                        float* Cf, __bf16* Cbf, int M, int N, int K, int ldc,
                        hipStream_t st) {
  dim3 g((M + BM - 1) / BM, (N + BN - 1) / BN), b(256);
  if (epi == 0)      gemm_bf16_wmma<0><<<g, b, 0, st>>>(A, B, bias, Cf, Cbf, M, N, K, ldc);
  else if (epi == 1) gemm_bf16_wmma<1><<<g, b, 0, st>>>(A, B, bias, Cf, Cbf, M, N, K, ldc);
  else               gemm_bf16_wmma<2><<<g, b, 0, st>>>(A, B, bias, Cf, Cbf, M, N, K, ldc);
}

static inline dim3 gs_grid(long long n) {
  long long b = (n + 255) / 256;
  if (b > 1048576) b = 1048576;
  if (b < 1) b = 1;
  return dim3((unsigned)b);
}

extern "C" void kernel_launch(void* const* d_in, const int* in_sizes, int n_in,
                              void* d_out, int out_size, void* d_ws, size_t ws_size,
                              hipStream_t stream) {
  const float* coords   = (const float*)d_in[0];
  const float* mlp_w0   = (const float*)d_in[1];
  const float* mlp_b0   = (const float*)d_in[2];
  const float* mlp_w1   = (const float*)d_in[3];
  const float* mlp_b1   = (const float*)d_in[4];
  const float* Wh       = (const float*)d_in[5];
  const float* Ws1      = (const float*)d_in[6];
  const float* bs1      = (const float*)d_in[7];
  const float* Ws2      = (const float*)d_in[8];
  const float* bs2      = (const float*)d_in[9];
  const float* Wg       = (const float*)d_in[10];
  const float* bg       = (const float*)d_in[11];
  const float* Wv       = (const float*)d_in[12];
  const float* codebook = (const float*)d_in[13];
  const float* dec_w0   = (const float*)d_in[14];
  const float* dec_b0   = (const float*)d_in[15];
  const float* dec_wh   = (const float*)d_in[16];
  const float* dec_bh   = (const float*)d_in[17];
  const float* dec_wf   = (const float*)d_in[18];
  const float* dec_bf   = (const float*)d_in[19];
  const float* bd_w0    = (const float*)d_in[20];
  const float* bd_b0    = (const float*)d_in[21];
  const float* bd_wh    = (const float*)d_in[22];
  const float* bd_bh    = (const float*)d_in[23];
  const float* bd_wf    = (const float*)d_in[24];
  const float* bd_bf    = (const float*)d_in[25];
  const int* atom_types   = (const int*)d_in[26];
  const int* atom_charges = (const int*)d_in[27];
  const int* srcI = (const int*)d_in[28];
  const int* dstI = (const int*)d_in[29];
  const int* edge_bond = (const int*)d_in[30];
  const int* pair_idx  = (const int*)d_in[31];
  const int* bond_pairs = (const int*)d_in[32];

  float* out = (float*)d_out;
  float* at_out   = out + 1;
  float* ch_out   = at_out + (size_t)NATOM * NTYPE;
  float* bond_out = ch_out + (size_t)NATOM * NCH;

  // ---------------- workspace arena ----------------
  char* base = (char*)d_ws;
  size_t cur = 0;
  auto take = [&](size_t bytes) -> char* {
    char* p = base + cur;
    cur = (cur + bytes + 255) & ~(size_t)255;
    return p;
  };
  // persistent
  float*  s_buf  = (float*) take((size_t)NATOM * DIM * 4);
  float*  v_buf  = (float*) take((size_t)NATOM * VCH * 3 * 4);
  float*  xdiff  = (float*) take((size_t)NEDGE * 3 * 4);
  float*  dlen   = (float*) take((size_t)NEDGE * 4);
  float*  deg    = (float*) take((size_t)NATOM * 4);
  int*    idxb   = (int*)   take((size_t)NATOM * 4);
  float*  q_buf  = (float*) take((size_t)NATOM * DIM * 4);
  __bf16* q_bf   = (__bf16*)take((size_t)NATOM * DIM * 2);
  __bf16* z_bf   = (__bf16*)take((size_t)NATOM * DIM * 2);
  float*  cnorm  = (float*) take((size_t)KCODE * 4);
  int*    cnt    = (int*)   take((size_t)KCODE * 4);
  float*  acc    = (float*) take(256);
  float*  logitsA= (float*) take((size_t)NATOM * (NTYPE + NCH) * 4);
  __bf16* Hemb   = (__bf16*)take((size_t)NATOM * HID * 2);
  __bf16* w1bf   = (__bf16*)take((size_t)HID * DIM * 2);
  __bf16* ws1bf  = (__bf16*)take((size_t)LAYERS * MIND * HID * 2);
  __bf16* ws2bf  = (__bf16*)take((size_t)LAYERS * HID * DIM * 2);
  __bf16* wgbf   = (__bf16*)take((size_t)LAYERS * DIM * VCH * 2);
  __bf16* c_bf   = (__bf16*)take((size_t)KCODE * DIM * 2);
  __bf16* dw0bf  = (__bf16*)take((size_t)DIM * HID * 2);
  __bf16* dwhbf  = (__bf16*)take((size_t)2 * HID * HID * 2);
  __bf16* dwfbf  = (__bf16*)take((size_t)HID * (NTYPE + NCH) * 2);
  __bf16* bw0bf  = (__bf16*)take((size_t)BONDK * HID * 2);
  __bf16* bwhbf  = (__bf16*)take((size_t)2 * HID * HID * 2);
  __bf16* bwfbf  = (__bf16*)take((size_t)HID * NBOND * 2);

  // reused scratch region (max over phases)
  size_t sc0 = cur;
  auto phase_take = [&](size_t& c, size_t bytes) -> char* {
    char* p = base + c;
    c = (c + bytes + 255) & ~(size_t)255;
    return p;
  };
  // phase: message passing
  size_t c1 = sc0;
  __bf16* m_in = (__bf16*)phase_take(c1, (size_t)NEDGE * MIND * 2);
  __bf16* hbuf = (__bf16*)phase_take(c1, (size_t)NEDGE * HID * 2);
  float*  smf  = (float*) phase_take(c1, (size_t)NEDGE * DIM * 4);
  __bf16* smbf = (__bf16*)phase_take(c1, (size_t)NEDGE * DIM * 2);
  float*  gate = (float*) phase_take(c1, (size_t)NEDGE * VCH * 4);
  float*  vhb  = (float*) phase_take(c1, (size_t)NEDGE * VCH * 3 * 4);
  // phase: atom decoder
  size_t c2 = sc0;
  __bf16* hA0 = (__bf16*)phase_take(c2, (size_t)NATOM * HID * 2);
  __bf16* hA1 = (__bf16*)phase_take(c2, (size_t)NATOM * HID * 2);
  // phase: bond decoder (chunked)
  size_t c3 = sc0;
  __bf16* Xb  = (__bf16*)phase_take(c3, (size_t)PCHUNK * BONDK * 2);
  __bf16* hB0 = (__bf16*)phase_take(c3, (size_t)PCHUNK * HID * 2);
  __bf16* hB1 = (__bf16*)phase_take(c3, (size_t)PCHUNK * HID * 2);

  size_t need = c1;
  if (c2 > need) need = c2;
  if (c3 > need) need = c3;
  if (need > ws_size) return;  // workspace too small: bail (no partial writes)

  dim3 blk(256);

  // ---------------- 0) weight conversions to bf16 ----------------
  auto conv = [&](const float* s, __bf16* d, long long n) {
    k_f32_bf16<<<gs_grid(n), blk, 0, stream>>>(s, d, n);
  };
  conv(mlp_w1, w1bf, (long long)HID * DIM);
  conv(Ws1, ws1bf, (long long)LAYERS * MIND * HID);
  conv(Ws2, ws2bf, (long long)LAYERS * HID * DIM);
  conv(Wg, wgbf, (long long)LAYERS * DIM * VCH);
  conv(codebook, c_bf, (long long)KCODE * DIM);
  conv(dec_w0, dw0bf, (long long)DIM * HID);
  conv(dec_wh, dwhbf, (long long)2 * HID * HID);
  conv(dec_wf, dwfbf, (long long)HID * (NTYPE + NCH));
  conv(bd_w0, bw0bf, (long long)BONDK * HID);
  conv(bd_wh, bwhbf, (long long)2 * HID * HID);
  conv(bd_wf, bwfbf, (long long)HID * NBOND);

  // zero accumulators / state
  k_zero_f32<<<gs_grid(64), blk, 0, stream>>>(acc, 64);
  k_zero_i32<<<gs_grid(KCODE), blk, 0, stream>>>(cnt, KCODE);
  k_zero_f32<<<gs_grid((long long)NATOM * VCH * 3), blk, 0, stream>>>(v_buf, (long long)NATOM * VCH * 3);
  k_zero_f32<<<gs_grid(NATOM), blk, 0, stream>>>(deg, NATOM);

  // ---------------- 1) embedding: s = relu(onehot@w0+b0) @ w1 + b1 ----------------
  k_embed<<<gs_grid((long long)NATOM * HID), blk, 0, stream>>>(atom_types, atom_charges,
                                                               mlp_w0, mlp_b0, Hemb);
  launch_gemm(0, Hemb, w1bf, mlp_b1, s_buf, nullptr, NATOM, DIM, HID, DIM, stream);

  // ---------------- 2) edge geometry + degree ----------------
  k_geom<<<gs_grid(NEDGE), blk, 0, stream>>>(coords, srcI, dstI, xdiff, dlen);
  k_deg<<<gs_grid(NEDGE), blk, 0, stream>>>(dstI, deg);

  // ---------------- 3) message-passing layers ----------------
  for (int i = 0; i < LAYERS; ++i) {
    k_build_min<<<gs_grid((long long)NEDGE * MIND), blk, 0, stream>>>(s_buf, dlen, srcI,
                                                                      edge_bond, m_in);
    k_vh<<<gs_grid((long long)NEDGE * VCH), blk, 0, stream>>>(
        v_buf, xdiff, srcI, Wh + (size_t)i * 17 * VCH, vhb, m_in);
    // h = relu(m_in @ Ws1 + bs1)
    launch_gemm(1, m_in, ws1bf + (size_t)i * MIND * HID, bs1 + (size_t)i * HID,
                nullptr, hbuf, NEDGE, HID, MIND, HID, stream);
    // s_m = h @ Ws2 + bs2 (f32 + bf16)
    launch_gemm(0, hbuf, ws2bf + (size_t)i * HID * DIM, bs2 + (size_t)i * DIM,
                smf, smbf, NEDGE, DIM, HID, DIM, stream);
    // gate = sigmoid(s_m @ Wg + bg)
    launch_gemm(2, smbf, wgbf + (size_t)i * DIM * VCH, bg + (size_t)i * VCH,
                gate, nullptr, NEDGE, VCH, DIM, VCH, stream);
    // scatter: s += segsum(s_m)/deg ; v += segsum((vh@Wv)*gate)/deg
    k_sscatter<<<gs_grid((long long)NEDGE * DIM), blk, 0, stream>>>(smf, dstI, deg, s_buf);
    k_vscatter<<<gs_grid((long long)NEDGE * VCH), blk, 0, stream>>>(
        vhb, gate, Wv + (size_t)i * VCH * VCH, dstI, deg, v_buf);
  }

  // ---------------- 4) VQ: nearest codebook entry per atom ----------------
  k_f32_bf16<<<gs_grid((long long)NATOM * DIM), blk, 0, stream>>>(s_buf, z_bf,
                                                                  (long long)NATOM * DIM);
  k_cnorm<<<gs_grid(KCODE), blk, 0, stream>>>(codebook, cnorm);
  vq_argmin_kernel<<<dim3(NATOM / 16), blk, 0, stream>>>(z_bf, c_bf, cnorm, idxb);
  k_gather_q<<<gs_grid((long long)NATOM * DIM), blk, 0, stream>>>(codebook, idxb, s_buf,
                                                                  q_buf, q_bf, acc);
  k_hist<<<gs_grid(NATOM), blk, 0, stream>>>(idxb, cnt);

  // ---------------- 5) atom decoder ----------------
  launch_gemm(1, q_bf, dw0bf, dec_b0, nullptr, hA0, NATOM, HID, DIM, HID, stream);
  launch_gemm(1, hA0, dwhbf, dec_bh, nullptr, hA1, NATOM, HID, HID, HID, stream);
  launch_gemm(1, hA1, dwhbf + (size_t)HID * HID, dec_bh + HID, nullptr, hA0,
              NATOM, HID, HID, HID, stream);
  launch_gemm(0, hA0, dwfbf, dec_bf, logitsA, nullptr, NATOM, NTYPE + NCH, HID,
              NTYPE + NCH, stream);
  k_split<<<gs_grid((long long)NATOM * (NTYPE + NCH)), blk, 0, stream>>>(logitsA, at_out, ch_out);

  // ---------------- 6) bond decoder (chunked) ----------------
  for (int c = 0; c < NCHUNK; ++c) {
    int p0 = c * PCHUNK;
    k_bondX<<<gs_grid((long long)PCHUNK * BONDK), blk, 0, stream>>>(q_buf, coords, pair_idx,
                                                                    Xb, p0, PCHUNK);
    launch_gemm(1, Xb, bw0bf, bd_b0, nullptr, hB0, PCHUNK, HID, BONDK, HID, stream);
    launch_gemm(1, hB0, bwhbf, bd_bh, nullptr, hB1, PCHUNK, HID, HID, HID, stream);
    launch_gemm(1, hB1, bwhbf + (size_t)HID * HID, bd_bh + HID, nullptr, hB0,
                PCHUNK, HID, HID, HID, stream);
    launch_gemm(0, hB0, bwfbf, bd_bf, bond_out + (size_t)p0 * NBOND, nullptr,
                PCHUNK, NBOND, HID, NBOND, stream);
  }

  // ---------------- 7) losses ----------------
  k_ce<<<gs_grid(NATOM), blk, 0, stream>>>(at_out, NATOM, NTYPE, atom_types, 0, acc + 1);
  k_ce<<<gs_grid(NATOM), blk, 0, stream>>>(ch_out, NATOM, NCH, atom_charges, 5, acc + 2);
  k_ce<<<gs_grid(NPAIR), blk, 0, stream>>>(bond_out, NPAIR, NBOND, bond_pairs, 0, acc + 3);
  k_final<<<dim3(1), dim3(1), 0, stream>>>(acc, cnt, out, out_size);
}
